// Stagate_18133351924036
// MI455X (gfx1250) — matrix operand order repaired and checked by reading, not compiled, not run on hardware
//
#include <hip/hip_runtime.h>
#include <hip/hip_bf16.h>
#include <math.h>

typedef __attribute__((ext_vector_type(16))) __bf16 v16bf;
typedef __attribute__((ext_vector_type(8)))  float  v8f;

static constexpr int IN_DIM = 3000;
static constexpr int HID    = 512;
static constexpr int OUTD   = 30;
static constexpr int KPAD1  = 3008;  // IN_DIM padded to multiple of 32 (GEMM1 K)
static constexpr int NPAD2  = 3072;  // IN_DIM padded to multiple of 512 (GEMM2 N)

// ---------- fp32 -> bf16 (round to nearest even) ----------
__device__ __forceinline__ __bf16 f2bf(float x) {
  union { float f; unsigned u; } c; c.f = x;
  unsigned r = c.u + 0x7FFFu + ((c.u >> 16) & 1u);
  unsigned short h = (unsigned short)(r >> 16);
  return __builtin_bit_cast(__bf16, h);
}

// ---------- W1 (3000x512 fp32) -> bf16 [KPAD1 x 512], zero padded ----------
__global__ void k_w1bf(const float* __restrict__ W1, __bf16* __restrict__ Wb) {
  int idx = blockIdx.x * blockDim.x + threadIdx.x;
  if (idx >= KPAD1 * HID) return;
  int k = idx / HID, n = idx % HID;
  Wb[idx] = (k < IN_DIM) ? f2bf(W1[(size_t)k * HID + n]) : f2bf(0.f);
}

// ---------- W1^T -> bf16 [512 x NPAD2], zero padded ----------
__global__ void k_w1tbf(const float* __restrict__ W1, __bf16* __restrict__ Wt) {
  int idx = blockIdx.x * blockDim.x + threadIdx.x;
  if (idx >= HID * NPAD2) return;
  int k = idx / NPAD2, n = idx % NPAD2;
  Wt[idx] = (n < IN_DIM) ? f2bf(W1[(size_t)n * HID + k]) : f2bf(0.f);
}

// ---------- WMMA fragment loads ----------
// A fragment (16x32 bf16): lane<16 row=lane&15, K chunks {+0..7, +16..23};
// lanes>=16 same rows, chunks {+8..15, +24..31}. `p` is already offset by
// row*lda + kb + hi*8, so chunks are p[0..7] and p[16..23].
__device__ __forceinline__ v16bf load_a_fast(const float* __restrict__ p) {
  float4 v0 = *(const float4*)(p);
  float4 v1 = *(const float4*)(p + 4);
  float4 v2 = *(const float4*)(p + 16);
  float4 v3 = *(const float4*)(p + 20);
  float f[16] = { v0.x, v0.y, v0.z, v0.w, v1.x, v1.y, v1.z, v1.w,
                  v2.x, v2.y, v2.z, v2.w, v3.x, v3.y, v3.z, v3.w };
  v16bf out;
#pragma unroll
  for (int i = 0; i < 16; ++i) out[i] = f2bf(f[i]);
  return out;
}

// Tail version: p points at row*lda (row base); element-guard k >= K with zero.
__device__ __forceinline__ v16bf load_a_tail(const float* __restrict__ p,
                                             int kb, int hi, int K) {
  int k1 = kb + hi * 8;
  int k2 = k1 + 16;
  float f[16];
#pragma unroll
  for (int i = 0; i < 8; ++i) {
    f[i]     = (k1 + i < K) ? p[k1 + i] : 0.f;
    f[8 + i] = (k2 + i < K) ? p[k2 + i] : 0.f;
  }
  v16bf out;
#pragma unroll
  for (int i = 0; i < 16; ++i) out[i] = f2bf(f[i]);
  return out;
}

// B fragment (32x16 bf16 from row-major [K x ldb] bf16): lane holds column N=nb+(lane&15),
// elements hold K rows {+0..7, +16..23} (lanes>=16: {+8..15, +24..31}), strided by ldb.
struct B32raw { float4 a, b; };
__device__ __forceinline__ v16bf load_b_frag(const __bf16* __restrict__ p, int ldb) {
  v16bf out;
#pragma unroll
  for (int i = 0; i < 8; ++i) {
    out[i]     = p[(size_t)i * ldb];
    out[8 + i] = p[(size_t)(16 + i) * ldb];
  }
  return out;
}

// ---------- bf16 WMMA GEMM: C[M x Nreal](fp32) = A[M x K](fp32) * B[K x ldb](bf16)
// block = 256 thr = 8 waves; wave tile 32M x 64N; block tile 32M x 512N ----------
__global__ __launch_bounds__(256) void k_gemm_bf16(
    const float* __restrict__ A, const __bf16* __restrict__ B, float* __restrict__ C,
    int M, int K, int ldb, int Nreal, int ldc) {
  const int wave = threadIdx.x >> 5;
  const int lane = threadIdx.x & 31;
  const int hi   = lane >> 4;
  const int m0   = blockIdx.x * 32;
  const int nb0  = blockIdx.y * 512 + wave * 64;

  int r0 = m0 + (lane & 15);      if (r0 >= M) r0 = 0;
  int r1 = m0 + 16 + (lane & 15); if (r1 >= M) r1 = 0;
  const float*  rowA0 = A + (size_t)r0 * K;                  // row base (for tail)
  const float*  rowA1 = A + (size_t)r1 * K;
  const float*  qa0   = rowA0 + hi * 8;                      // fast-path cursor
  const float*  qa1   = rowA1 + hi * 8;
  const __bf16* qb    = B + (size_t)(hi * 8) * ldb + nb0 + (lane & 15); // B column cursor
  const size_t  bstep = (size_t)32 * ldb;

  v8f acc[2][4];
#pragma unroll
  for (int i = 0; i < 2; ++i)
#pragma unroll
    for (int j = 0; j < 4; ++j) acc[i][j] = (v8f)(0.f);

  const int kfull = K & ~31;
  for (int kb = 0; kb < kfull; kb += 32) {
    v16bf a0 = load_a_fast(qa0);
    v16bf a1 = load_a_fast(qa1);
    v16bf b0 = load_b_frag(qb, ldb);
    v16bf b1 = load_b_frag(qb + 16, ldb);
    v16bf b2 = load_b_frag(qb + 32, ldb);
    v16bf b3 = load_b_frag(qb + 48, ldb);
    __builtin_prefetch(qa0 + 128, 0, 1);   // global_prefetch_b8: stream A ahead
    qa0 += 32; qa1 += 32; qb += bstep;
    acc[0][0] = __builtin_amdgcn_wmma_f32_16x16x32_bf16(false, a0, false, b0, (short)0, acc[0][0], false, false);
    acc[0][1] = __builtin_amdgcn_wmma_f32_16x16x32_bf16(false, a0, false, b1, (short)0, acc[0][1], false, false);
    acc[0][2] = __builtin_amdgcn_wmma_f32_16x16x32_bf16(false, a0, false, b2, (short)0, acc[0][2], false, false);
    acc[0][3] = __builtin_amdgcn_wmma_f32_16x16x32_bf16(false, a0, false, b3, (short)0, acc[0][3], false, false);
    acc[1][0] = __builtin_amdgcn_wmma_f32_16x16x32_bf16(false, a1, false, b0, (short)0, acc[1][0], false, false);
    acc[1][1] = __builtin_amdgcn_wmma_f32_16x16x32_bf16(false, a1, false, b1, (short)0, acc[1][1], false, false);
    acc[1][2] = __builtin_amdgcn_wmma_f32_16x16x32_bf16(false, a1, false, b2, (short)0, acc[1][2], false, false);
    acc[1][3] = __builtin_amdgcn_wmma_f32_16x16x32_bf16(false, a1, false, b3, (short)0, acc[1][3], false, false);
  }

  if (kfull < K) {                       // peeled K tail (A guarded; B zero-padded)
    v16bf a0 = load_a_tail(rowA0, kfull, hi, K);
    v16bf a1 = load_a_tail(rowA1, kfull, hi, K);
    v16bf b0 = load_b_frag(qb, ldb);
    v16bf b1 = load_b_frag(qb + 16, ldb);
    v16bf b2 = load_b_frag(qb + 32, ldb);
    v16bf b3 = load_b_frag(qb + 48, ldb);
    acc[0][0] = __builtin_amdgcn_wmma_f32_16x16x32_bf16(false, a0, false, b0, (short)0, acc[0][0], false, false);
    acc[0][1] = __builtin_amdgcn_wmma_f32_16x16x32_bf16(false, a0, false, b1, (short)0, acc[0][1], false, false);
    acc[0][2] = __builtin_amdgcn_wmma_f32_16x16x32_bf16(false, a0, false, b2, (short)0, acc[0][2], false, false);
    acc[0][3] = __builtin_amdgcn_wmma_f32_16x16x32_bf16(false, a0, false, b3, (short)0, acc[0][3], false, false);
    acc[1][0] = __builtin_amdgcn_wmma_f32_16x16x32_bf16(false, a1, false, b0, (short)0, acc[1][0], false, false);
    acc[1][1] = __builtin_amdgcn_wmma_f32_16x16x32_bf16(false, a1, false, b1, (short)0, acc[1][1], false, false);
    acc[1][2] = __builtin_amdgcn_wmma_f32_16x16x32_bf16(false, a1, false, b2, (short)0, acc[1][2], false, false);
    acc[1][3] = __builtin_amdgcn_wmma_f32_16x16x32_bf16(false, a1, false, b3, (short)0, acc[1][3], false, false);
  }

  // C/D layout: lane&15 -> N, VGPR v -> M = v (+8 for lanes 16..31)
  const int rowSel = hi * 8;
#pragma unroll
  for (int i = 0; i < 2; ++i) {
#pragma unroll
    for (int j = 0; j < 4; ++j) {
      int col = nb0 + j * 16 + (lane & 15);
      if (col < Nreal) {
        int mb = m0 + i * 16 + rowSel;
#pragma unroll
        for (int v = 0; v < 8; ++v) {
          int mr = mb + v;
          if (mr < M) C[(size_t)mr * ldc + col] = acc[i][j][v];
        }
      }
    }
  }
}

// ---------- attention scores: a_src = x1 @ att_s, a_dst = x1 @ att_d ----------
__global__ void k_att(const float* __restrict__ x1, const float* __restrict__ att_s,
                      const float* __restrict__ att_d, float* __restrict__ as,
                      float* __restrict__ ad, int N) {
  int lane = threadIdx.x & 31;
  int node = blockIdx.x * 8 + (threadIdx.x >> 5);
  if (node >= N) return;
  const float* r = x1 + (size_t)node * HID;
  float s = 0.f, d = 0.f;
  for (int c = lane; c < HID; c += 32) {
    float v = r[c];
    s += v * att_s[c];
    d += v * att_d[c];
  }
#pragma unroll
  for (int off = 16; off > 0; off >>= 1) {
    s += __shfl_xor(s, off, 32);
    d += __shfl_xor(d, off, 32);
  }
  if (lane == 0) { as[node] = s; ad[node] = d; }
}

// ---------- per-edge sigmoid + segment max (positive floats as uint) ----------
__global__ void k_edge_e(const int* __restrict__ src, const int* __restrict__ dst,
                         const float* __restrict__ as, const float* __restrict__ ad,
                         float* __restrict__ e, unsigned* __restrict__ segmax, int E) {
  int i = blockIdx.x * blockDim.x + threadIdx.x;
  if (i >= E) return;
  float v = as[src[i]] + ad[dst[i]];
  float s = 1.f / (1.f + __expf(-v));   // sigmoid in (0,1): uint-monotone
  e[i] = s;
  atomicMax(&segmax[dst[i]], __float_as_uint(s));
}

// ---------- exp(e - max) + segment sum ----------
__global__ void k_edge_exp(const int* __restrict__ dst, float* __restrict__ e,
                           const float* __restrict__ segmax, float* __restrict__ segsum,
                           int E) {
  int i = blockIdx.x * blockDim.x + threadIdx.x;
  if (i >= E) return;
  int d = dst[i];
  float ex = __expf(e[i] - segmax[d]);
  e[i] = ex;
  atomicAdd(&segsum[d], ex);
}

// ---------- aggregate: H[dst] += X[src] * (ex / (segsum+eps)); wave per edge ----------
__global__ void k_aggregate(const int* __restrict__ src, const int* __restrict__ dst,
                            const float* __restrict__ ex, const float* __restrict__ segsum,
                            const float* __restrict__ X, float* __restrict__ H, int E) {
  int lane = threadIdx.x & 31;
  int edge = blockIdx.x * 8 + (threadIdx.x >> 5);
  if (edge >= E) return;
  int s = src[edge], d = dst[edge];
  float alpha = ex[edge] / (segsum[d] + 1e-16f);
  const float* xs = X + (size_t)s * HID;
  float* hd = H + (size_t)d * HID;
  for (int c = lane; c < HID; c += 32) atomicAdd(&hd[c], xs[c] * alpha);
}

// ---------- ELU in place ----------
__global__ void k_elu(float* __restrict__ h, int n) {
  int i = blockIdx.x * blockDim.x + threadIdx.x;
  if (i >= n) return;
  float x = h[i];
  h[i] = x > 0.f ? x : expm1f(x);
}

// ---------- h2 = h1 @ W2 (512x30); lane = out column ----------
__global__ void k_h2(const float* __restrict__ h1, const float* __restrict__ W2,
                     float* __restrict__ h2, int N) {
  int lane = threadIdx.x & 31;
  int node = blockIdx.x * 8 + (threadIdx.x >> 5);
  if (node >= N || lane >= OUTD) return;
  const float* r = h1 + (size_t)node * HID;
  float acc = 0.f;
  for (int k = 0; k < HID; ++k) acc += r[k] * W2[k * OUTD + lane];
  h2[(size_t)node * OUTD + lane] = acc;
}

// ---------- x3 = h2 @ W2^T  (x3[n,c] = sum_k h2[n,k] * W2[c,k]) ----------
__global__ void k_x3(const float* __restrict__ h2, const float* __restrict__ W2,
                     float* __restrict__ x3, int N) {
  int c    = (int)threadIdx.x + (int)(blockIdx.x & 1) * 256;
  int node = (int)(blockIdx.x >> 1);
  if (node >= N) return;
  const float* r = h2 + (size_t)node * OUTD;
  const float* w = W2 + (size_t)c * OUTD;
  float acc = 0.f;
#pragma unroll
  for (int k = 0; k < OUTD; ++k) acc += r[k] * w[k];
  x3[(size_t)node * HID + c] = acc;
}

extern "C" void kernel_launch(void* const* d_in, const int* in_sizes, int n_in,
                              void* d_out, int out_size, void* d_ws, size_t ws_size,
                              hipStream_t stream) {
  (void)n_in; (void)out_size; (void)ws_size;
  const float* features = (const float*)d_in[0];
  const int*   ei       = (const int*)d_in[1];
  const float* W1       = (const float*)d_in[2];
  const float* W2       = (const float*)d_in[3];
  const float* att_s    = (const float*)d_in[4];
  const float* att_d    = (const float*)d_in[5];

  const int N = in_sizes[0] / IN_DIM;   // 20000
  const int E = in_sizes[1] / 2;        // 320000
  const int* src = ei;
  const int* dst = ei + E;

  float* out_h2 = (float*)d_out;
  float* out_h4 = (float*)d_out + (size_t)N * OUTD;

  // workspace carve (256B aligned)
  char* w = (char*)d_ws;
  auto carve = [&](size_t bytes) -> void* {
    void* p = (void*)w;
    w += (bytes + 255) & ~(size_t)255;
    return p;
  };
  __bf16* w1bf   = (__bf16*)carve((size_t)KPAD1 * HID * 2);
  __bf16* w1tbf  = (__bf16*)carve((size_t)HID * NPAD2 * 2);
  float*  x1     = (float*)carve((size_t)N * HID * 4);
  float*  h1     = (float*)carve((size_t)N * HID * 4);
  float*  asrc   = (float*)carve((size_t)N * 4);
  float*  adst   = (float*)carve((size_t)N * 4);
  float*  ebuf   = (float*)carve((size_t)E * 4);
  float*  segmax = (float*)carve((size_t)N * 4);
  float*  segsum = (float*)carve((size_t)N * 4);
  float*  x3 = x1;  // x1 dead once aggregate1 done; x3 produced after
  float*  h3 = h1;  // h1 dead once h2 computed; re-zeroed below

  // zero accumulators (every call: launch must be replay-deterministic)
  hipMemsetAsync(segmax, 0, (size_t)N * 4, stream);
  hipMemsetAsync(segsum, 0, (size_t)N * 4, stream);
  hipMemsetAsync(h1, 0, (size_t)N * HID * 4, stream);

  // bf16 weight prep (B operands, live in L2)
  k_w1bf <<<(KPAD1 * HID + 255) / 256, 256, 0, stream>>>(W1, w1bf);
  k_w1tbf<<<(HID * NPAD2 + 255) / 256, 256, 0, stream>>>(W1, w1tbf);

  // conv1 linear: x1 = features @ W1
  dim3 g1((N + 31) / 32, 1);
  k_gemm_bf16<<<g1, 256, 0, stream>>>(features, w1bf, x1, N, IN_DIM, HID, HID, HID);

  // attention (shared by conv1 and conv3)
  k_att<<<(N + 7) / 8, 256, 0, stream>>>(x1, att_s, att_d, asrc, adst, N);
  k_edge_e <<<(E + 255) / 256, 256, 0, stream>>>(src, dst, asrc, adst, ebuf,
                                                 (unsigned*)segmax, E);
  k_edge_exp<<<(E + 255) / 256, 256, 0, stream>>>(dst, ebuf, segmax, segsum, E);

  // conv1 propagate + ELU
  k_aggregate<<<(E + 7) / 8, 256, 0, stream>>>(src, dst, ebuf, segsum, x1, h1, E);
  k_elu<<<(N * HID + 255) / 256, 256, 0, stream>>>(h1, N * HID);

  // conv2: h2 = h1 @ W2 (written straight to output)
  k_h2<<<(N + 7) / 8, 256, 0, stream>>>(h1, W2, out_h2, N);

  // conv3: x3 = h2 @ W2^T, propagate (same alpha), ELU
  hipMemsetAsync(h3, 0, (size_t)N * HID * 4, stream);
  k_x3<<<N * 2, 256, 0, stream>>>(out_h2, W2, x3, N);
  k_aggregate<<<(E + 7) / 8, 256, 0, stream>>>(src, dst, ebuf, segsum, x3, h3, E);
  k_elu<<<(N * HID + 255) / 256, 256, 0, stream>>>(h3, N * HID);

  // conv4: h4 = h3 @ W1^T
  dim3 g2((N + 31) / 32, NPAD2 / 512);
  k_gemm_bf16<<<g2, 256, 0, stream>>>(h3, w1tbf, out_h4, N, HID, NPAD2, IN_DIM, IN_DIM);
}
